// VRWKV_WKV_ChannelMix_12249246728672
// MI455X (gfx1250) — compile-verified, matrix operand order checked by reading, not run on hardware
//
#include <hip/hip_runtime.h>
#include <hip/hip_bf16.h>

typedef __attribute__((ext_vector_type(16))) _Float16 v16h;
typedef __attribute__((ext_vector_type(8)))  _Float16 v8h;
typedef __attribute__((ext_vector_type(8)))  float    v8f;

#define BB 16
#define TT 4096
#define CC 256
#define HW 64
#define MM (BB * TT)        // 65536 rows
#define NCHUNK 32
#define CLEN (TT / NCHUNK)  // 128 steps per chunk

// ---------------------------------------------------------------------------
// fp32 -> f16 weight conversion
// ---------------------------------------------------------------------------
__global__ void cvt_f32_f16(const float* __restrict__ in, _Float16* __restrict__ out, int n) {
  int i = blockIdx.x * blockDim.x + threadIdx.x;
  if (i < n) out[i] = (_Float16)in[i];
}

// ---------------------------------------------------------------------------
// OmniShift: alpha0*I + alpha1*dw1x1 + alpha2*dw3x3 + alpha3*dw5x5 on the
// NCHW view of x[B,T,C]; combined into one 5x5 per-channel kernel held in
// registers. Block = (b, row y), threads = 256 channels -> coalesced IO.
// Output written directly as f16 (WMMA A operand for the GEMMs).
// ---------------------------------------------------------------------------
__global__ void omnishift_kernel(const float* __restrict__ x,
                                 const float* __restrict__ alpha,
                                 const float* __restrict__ w1,
                                 const float* __restrict__ w3,
                                 const float* __restrict__ w5,
                                 _Float16* __restrict__ xf16) {
  const int c  = threadIdx.x;       // 0..255 (channel)
  const int by = blockIdx.x;        // b*64 + y
  const int b  = by >> 6;
  const int yy = by & 63;

  const float a0 = alpha[0], a1 = alpha[1], a2 = alpha[2], a3 = alpha[3];

  float kw[25];
#pragma unroll
  for (int j = 0; j < 25; ++j) kw[j] = a3 * w5[c * 25 + j];
#pragma unroll
  for (int j = 0; j < 3; ++j)
#pragma unroll
    for (int i = 0; i < 3; ++i)
      kw[(j + 1) * 5 + (i + 1)] += a2 * w3[c * 9 + j * 3 + i];
  kw[12] += a1 * w1[c] + a0;

  const float* xb = x + (size_t)b * TT * CC;

  for (int xx = 0; xx < HW; ++xx) {
    float s = 0.f;
#pragma unroll
    for (int dy = -2; dy <= 2; ++dy) {
      const int y2 = yy + dy;
      if ((unsigned)y2 >= (unsigned)HW) continue;
#pragma unroll
      for (int dx = -2; dx <= 2; ++dx) {
        const int x2 = xx + dx;
        if ((unsigned)x2 >= (unsigned)HW) continue;
        s += kw[(dy + 2) * 5 + (dx + 2)] * xb[(size_t)(y2 * HW + x2) * CC + c];
      }
    }
    xf16[((size_t)b * TT + yy * HW + xx) * CC + c] = (_Float16)s;
  }
}

// ---------------------------------------------------------------------------
// WMMA fragment loader for 16-bit 16x32 A (and B^T stored [N,K] row-major,
// which has the identical per-lane pattern). ISA per-lane K order:
//   lanes 0-15 : VGPR0-3 = K 0..7,  VGPR4-7 = K 16..23
//   lanes 16-31: VGPR0-3 = K 8..15, VGPR4-7 = K 24..31
// -> two contiguous 16B loads per lane.
// ---------------------------------------------------------------------------
__device__ __forceinline__ v16h load_frag16(const _Float16* __restrict__ base,
                                            int row, int k0, int half) {
  const _Float16* p = base + (size_t)row * CC + k0 + half * 8;
  v8h lo = *(const v8h*)(p);
  v8h hi = *(const v8h*)(p + 16);
  v16h r;
#pragma unroll
  for (int i = 0; i < 8; ++i) { r[i] = lo[i]; r[i + 8] = hi[i]; }
  return r;
}

// ---------------------------------------------------------------------------
// Fused k/v GEMM: k = relu(xf@Wk^T)^2 (fp32 out), v = xf@Wv^T (fp32 out).
// One A fragment feeds two WMMAs. 8 waves/block -> 64x32 block tile.
// ---------------------------------------------------------------------------
__global__ void gemm_kv_wmma(const _Float16* __restrict__ A,   // xf16 [M, K=256]
                             const _Float16* __restrict__ Bk,  // Wk16 [N=256, K=256]
                             const _Float16* __restrict__ Bv,  // Wv16 [N=256, K=256]
                             float* __restrict__ Kout,
                             float* __restrict__ Vout) {
  const int lane = threadIdx.x & 31;
  const int wv   = threadIdx.x >> 5;                 // 0..7
  const int m0   = blockIdx.x * 64 + (wv & 3) * 16;
  const int n0   = blockIdx.y * 32 + (wv >> 2) * 16;
  const int r    = lane & 15;
  const int half = lane >> 4;

  v8f ck = {}, cv = {};
#pragma unroll
  for (int k0 = 0; k0 < CC; k0 += 32) {
    v16h a  = load_frag16(A,  m0 + r, k0, half);
    v16h bk = load_frag16(Bk, n0 + r, k0, half);
    v16h bv = load_frag16(Bv, n0 + r, k0, half);
    ck = __builtin_amdgcn_wmma_f32_16x16x32_f16(false, a, false, bk, (short)0, ck, false, false);
    cv = __builtin_amdgcn_wmma_f32_16x16x32_f16(false, a, false, bv, (short)0, cv, false, false);
  }

  const int n = n0 + r;  // D layout: lane = column, VGPR i = row (+8 for hi half)
#pragma unroll
  for (int i = 0; i < 8; ++i) {
    const int m = m0 + i + half * 8;
    float kk = ck[i];
    kk = kk > 0.f ? kk : 0.f;
    Kout[(size_t)m * CC + n] = kk * kk;
    Vout[(size_t)m * CC + n] = cv[i];
  }
}

// ---------------------------------------------------------------------------
// WKV as an associative chunked scan. A span of steps is the transform
//   o' = max(W+o, O); p' = e^{W+o-o'} p + e^{O-o'} P; q' = e^{W+o-o'} q + e^{O-o'} Q
// with single-step (W=w, O=k_t, P=v_t, Q=1). Phase 1 folds each 128-step
// chunk; phase 2 scans the 32 chunk transforms per (b,c) to get the state
// entering each chunk; phase 3 replays chunks from those states to emit y.
// Serial depth: 4096 -> 128+32+128.
// ---------------------------------------------------------------------------
__global__ void wkv_chunk_transform(const float* __restrict__ K, const float* __restrict__ V,
                                    const float* __restrict__ sdecay,
                                    float* __restrict__ Wc, float* __restrict__ Oc,
                                    float* __restrict__ Pc, float* __restrict__ Qc) {
  const int c = threadIdx.x;                // channel (coalesced)
  const int j = blockIdx.x & (NCHUNK - 1);  // chunk
  const int b = blockIdx.x >> 5;            // batch
  const float w = sdecay[c] / (float)TT;

  float Wt = 0.f, Ot = -1e38f, Pt = 0.f, Qt = 0.f;
  size_t idx = ((size_t)b * TT + (size_t)j * CLEN) * CC + c;
  for (int s = 0; s < CLEN; ++s, idx += CC) {
    const float kt = K[idx], vt = V[idx];
    // T_new = T_step o T_acc
    const float On = fmaxf(w + Ot, kt);
    const float ea = __expf(w + Ot - On);
    const float eb = __expf(kt - On);
    Pt = ea * Pt + eb * vt;
    Qt = ea * Qt + eb;
    Ot = On;
    Wt += w;
  }
  const size_t o = ((size_t)b * NCHUNK + j) * CC + c;
  Wc[o] = Wt; Oc[o] = Ot; Pc[o] = Pt; Qc[o] = Qt;
}

__global__ void wkv_chunk_scan(const float* __restrict__ Wc, const float* __restrict__ Oc,
                               const float* __restrict__ Pc, const float* __restrict__ Qc,
                               float* __restrict__ Sp, float* __restrict__ Sq,
                               float* __restrict__ So) {
  const int c = threadIdx.x;
  const int b = blockIdx.x;
  float p = 0.f, q = 0.f, o = -1e38f;
  for (int j = 0; j < NCHUNK; ++j) {
    const size_t idx = ((size_t)b * NCHUNK + j) * CC + c;
    Sp[idx] = p; Sq[idx] = q; So[idx] = o;   // state entering chunk j
    const float W = Wc[idx], O = Oc[idx], P = Pc[idx], Q = Qc[idx];
    const float on = fmaxf(W + o, O);
    const float ea = __expf(W + o - on);
    const float eb = __expf(O - on);
    p = ea * p + eb * P;
    q = ea * q + eb * Q;
    o = on;
  }
}

__global__ void wkv_chunk_emit(const float* __restrict__ K, const float* __restrict__ V,
                               const float* __restrict__ sdecay, const float* __restrict__ sfirst,
                               const float* __restrict__ Sp, const float* __restrict__ Sq,
                               const float* __restrict__ So,
                               _Float16* __restrict__ VV) {
  const int c = threadIdx.x;
  const int j = blockIdx.x & (NCHUNK - 1);
  const int b = blockIdx.x >> 5;
  const float w = sdecay[c] / (float)TT;
  const float u = sfirst[c] / (float)TT;

  const size_t sidx = ((size_t)b * NCHUNK + j) * CC + c;
  float p = Sp[sidx], q = Sq[sidx], o = So[sidx];

  size_t idx = ((size_t)b * TT + (size_t)j * CLEN) * CC + c;
  for (int s = 0; s < CLEN; ++s, idx += CC) {
    const float kt = K[idx], vt = V[idx];
    const float no = fmaxf(o, u + kt);
    const float Aa = __expf(o - no);
    const float Bc = __expf(u + kt - no);
    VV[idx] = (_Float16)((Aa * p + Bc * vt) / (Aa * q + Bc));
    const float no2 = fmaxf(w + o, kt);
    const float A2 = __expf(w + o - no2);
    const float B2 = __expf(kt - no2);
    p = A2 * p + B2 * vt;
    q = A2 * q + B2;
    o = no2;
  }
}

// ---------------------------------------------------------------------------
// Fused r-gate + output GEMM: out = sigmoid(xf@Wr^T) * (vv@Wo^T), fp32 out.
// ---------------------------------------------------------------------------
__global__ void gemm_out_wmma(const _Float16* __restrict__ Axf,  // xf16 [M, 256]
                              const _Float16* __restrict__ Avv,  // vv16 [M, 256]
                              const _Float16* __restrict__ Br,   // Wr16 [256, 256]
                              const _Float16* __restrict__ Bo,   // Wo16 [256, 256]
                              float* __restrict__ Out) {
  const int lane = threadIdx.x & 31;
  const int wv   = threadIdx.x >> 5;
  const int m0   = blockIdx.x * 64 + (wv & 3) * 16;
  const int n0   = blockIdx.y * 32 + (wv >> 2) * 16;
  const int r    = lane & 15;
  const int half = lane >> 4;

  v8f cr = {}, co = {};
#pragma unroll
  for (int k0 = 0; k0 < CC; k0 += 32) {
    v16h ar = load_frag16(Axf, m0 + r, k0, half);
    v16h av = load_frag16(Avv, m0 + r, k0, half);
    v16h br = load_frag16(Br,  n0 + r, k0, half);
    v16h bo = load_frag16(Bo,  n0 + r, k0, half);
    cr = __builtin_amdgcn_wmma_f32_16x16x32_f16(false, ar, false, br, (short)0, cr, false, false);
    co = __builtin_amdgcn_wmma_f32_16x16x32_f16(false, av, false, bo, (short)0, co, false, false);
  }

  const int n = n0 + r;
#pragma unroll
  for (int i = 0; i < 8; ++i) {
    const int m = m0 + i + half * 8;
    const float gate = 1.f / (1.f + __expf(-cr[i]));
    Out[(size_t)m * CC + n] = gate * co[i];
  }
}

// ---------------------------------------------------------------------------
// Launch
// ---------------------------------------------------------------------------
extern "C" void kernel_launch(void* const* d_in, const int* in_sizes, int n_in,
                              void* d_out, int out_size, void* d_ws, size_t ws_size,
                              hipStream_t stream) {
  const float* x      = (const float*)d_in[0];
  const float* alpha  = (const float*)d_in[1];
  const float* w1     = (const float*)d_in[2];
  const float* w3     = (const float*)d_in[3];
  const float* w5     = (const float*)d_in[4];
  const float* Wk     = (const float*)d_in[5];
  const float* Wr     = (const float*)d_in[6];
  const float* Wv     = (const float*)d_in[7];
  const float* Wo     = (const float*)d_in[8];
  const float* sdecay = (const float*)d_in[9];
  const float* sfirst = (const float*)d_in[10];

  char* ws = (char*)d_ws;
  const size_t NW = (size_t)CC * CC;                   // 65536 weight elements
  _Float16* Wk16 = (_Float16*)ws;
  _Float16* Wr16 = Wk16 + NW;
  _Float16* Wv16 = Wr16 + NW;
  _Float16* Wo16 = Wv16 + NW;
  _Float16* xf16 = Wo16 + NW;                          // 512K .. +32MB
  float*    k32  = (float*)(ws + 524288 + (size_t)MM * CC * 2);     // +64MB
  float*    v32  = (float*)((char*)k32 + (size_t)MM * CC * 4);      // +64MB
  _Float16* vv16 = (_Float16*)((char*)v32 + (size_t)MM * CC * 4);   // +32MB
  // chunked-scan scratch: 7 arrays of B*NCHUNK*C floats (512KB each)
  float* Wc = (float*)((char*)vv16 + (size_t)MM * CC * 2);
  const size_t NT = (size_t)BB * NCHUNK * CC;          // 131072
  float* Oc = Wc + NT;
  float* Pc = Oc + NT;
  float* Qc = Pc + NT;
  float* Sp = Qc + NT;
  float* Sq = Sp + NT;
  float* So = Sq + NT;

  // 1) weights -> f16
  {
    dim3 g((unsigned)((NW + 255) / 256)), blk(256);
    cvt_f32_f16<<<g, blk, 0, stream>>>(Wk, Wk16, (int)NW);
    cvt_f32_f16<<<g, blk, 0, stream>>>(Wr, Wr16, (int)NW);
    cvt_f32_f16<<<g, blk, 0, stream>>>(Wv, Wv16, (int)NW);
    cvt_f32_f16<<<g, blk, 0, stream>>>(Wo, Wo16, (int)NW);
  }

  // 2) OmniShift -> xf16
  omnishift_kernel<<<dim3(BB * HW), dim3(CC), 0, stream>>>(x, alpha, w1, w3, w5, xf16);

  // 3) k/v GEMMs (WMMA)
  gemm_kv_wmma<<<dim3(MM / 64, CC / 32), dim3(256), 0, stream>>>(xf16, Wk16, Wv16, k32, v32);

  // 4) WKV chunked parallel scan -> vv16
  wkv_chunk_transform<<<dim3(BB * NCHUNK), dim3(CC), 0, stream>>>(k32, v32, sdecay,
                                                                  Wc, Oc, Pc, Qc);
  wkv_chunk_scan<<<dim3(BB), dim3(CC), 0, stream>>>(Wc, Oc, Pc, Qc, Sp, Sq, So);
  wkv_chunk_emit<<<dim3(BB * NCHUNK), dim3(CC), 0, stream>>>(k32, v32, sdecay, sfirst,
                                                             Sp, Sq, So, vv16);

  // 5) fused gate + output GEMM (WMMA) -> d_out
  gemm_out_wmma<<<dim3(MM / 64, CC / 32), dim3(256), 0, stream>>>(
      xf16, vv16, Wr16, Wo16, (float*)d_out);
}
